// MSARowAttention_79345225826825
// MI455X (gfx1250) — matrix operand order repaired and checked by reading, not compiled, not run on hardware
//
#include <hip/hip_runtime.h>

typedef __attribute__((ext_vector_type(2)))  float  v2f;
typedef __attribute__((ext_vector_type(4)))  float  v4f;
typedef __attribute__((ext_vector_type(8)))  float  v8f;
typedef __attribute__((ext_vector_type(8)))  __bf16 v8bf;
typedef __attribute__((ext_vector_type(16))) __bf16 v16bf;

constexpr int NBATCH = 2;
constexpr int SEQ    = 768;
constexpr int EMB    = 512;
constexpr int NH     = 8;
constexpr int HD     = 64;
constexpr int PD     = 128;
constexpr int NROW   = NBATCH * SEQ;   // 1536 GEMM rows
constexpr int KSPLIT = 4;              // key-dim split in attention
constexpr int KSEG   = SEQ / KSPLIT;   // 192 keys per split

// ---------------- WMMA fragment helpers (wave32, gfx1250 layouts) ----------------
__device__ __forceinline__ v16bf cat16(v8bf a, v8bf b) {
  return __builtin_shufflevector(a, b, 0,1,2,3,4,5,6,7,8,9,10,11,12,13,14,15);
}
// A-matrix 16x32 bf16: lane L holds row M=L%16; K = hi*16 + 0..15 contiguous
// (hi = L/16). Row must be K-contiguous.
__device__ __forceinline__ v16bf load_a16(const __bf16* __restrict__ row, int kc, int hi) {
  v8bf lo = *(const v8bf*)(row + kc + hi * 16);
  v8bf hb = *(const v8bf*)(row + kc + hi * 16 + 8);
  return cat16(lo, hb);
}
// B-matrix 32x16 bf16: lane L holds col N=L%16; K = hi*16 + 0..15 contiguous.
__device__ __forceinline__ v16bf load_b16(const __bf16* __restrict__ row, int kc, int hi) {
  v8bf lo = *(const v8bf*)(row + kc + hi * 16);
  v8bf hb = *(const v8bf*)(row + kc + hi * 16 + 8);
  return cat16(lo, hb);
}
__device__ __forceinline__ v8f wmma_bf16(v16bf a, v16bf b, v8f c) {
  return __builtin_amdgcn_wmma_f32_16x16x32_bf16(false, a, false, b, (short)0, c, false, false);
}

// ---------------- Kernel 1: convert x / W to bf16, build padded Wp/bp ----------------
__global__ __launch_bounds__(256) void convert_kernel(
    const float* __restrict__ x, const float* __restrict__ Wq,
    const float* __restrict__ Wk, const float* __restrict__ Wv,
    const float* __restrict__ Wo, const float* __restrict__ Wp,
    const float* __restrict__ bp,
    __bf16* __restrict__ xb, __bf16* __restrict__ wqkv, __bf16* __restrict__ wob,
    float* __restrict__ wpm, float* __restrict__ bpad) {
  long long t = (long long)blockIdx.x * 256 + threadIdx.x;
  const long long NX = (long long)NROW * EMB;   // 786432
  const long long NW = (long long)EMB * EMB;    // 262144
  if (t < NX) { xb[t] = (__bf16)x[t]; return; }
  t -= NX;
  if (t < NW) { wqkv[t] = (__bf16)Wq[t]; return; }
  t -= NW;
  if (t < NW) { wqkv[NW + t] = (__bf16)Wk[t]; return; }
  t -= NW;
  if (t < NW) { wqkv[2 * NW + t] = (__bf16)Wv[t]; return; }
  t -= NW;
  if (t < NW) { wob[t] = (__bf16)Wo[t]; return; }
  t -= NW;
  if (t < 16 * PD) {               // Wp padded to 16 rows (rows 8..15 zero)
    const int n = (int)(t >> 7), col = (int)(t & (PD - 1));
    wpm[t] = (n < NH) ? Wp[n * PD + col] : 0.f;
    return;
  }
  t -= 16 * PD;
  if (t < 16) bpad[t] = (t < NH) ? bp[t] : 0.f;
}

// ---------------- Kernel 2: pair bias  pb[b,h,i,j] = pair[b,i,j,:]·Wp[h,:] + bp[h] ----
// One wave per 16 j's (rows M of a 16x16 f32 WMMA tile, heads on N), K = 128 in 32 steps.
// Inner loop is pure load->v_wmma_f32_16x16x4_f32 (Wp pre-masked, bias pre-padded).
__global__ __launch_bounds__(256) void pair_bias_kernel(
    const float* __restrict__ pair, const float* __restrict__ wpm,
    const float* __restrict__ bpad, float* __restrict__ pb) {
  const int lane = threadIdx.x & 31;
  const int wave = threadIdx.x >> 5;
  const int l16 = lane & 15, hi = lane >> 4;
  const int bi = blockIdx.x / (SEQ / 128);     // b*SEQ + i
  const int jg = blockIdx.x % (SEQ / 128);
  const int b  = bi / SEQ, i = bi % SEQ;
  const int j0 = jg * 128 + wave * 16;

  // A-matrix f32 16x4: lane holds row M=l16; VGPR0 K=k0+2*hi, VGPR1 K=k0+2*hi+1
  const float* ap   = pair + ((long long)bi * SEQ + j0 + l16) * PD + 2 * hi;
  const float* wrow = wpm + l16 * PD + 2 * hi;   // B[k][n] = wpm[n][k], rows >=8 are zero
  const float  bias = bpad[l16];

  v8f acc;
  #pragma unroll
  for (int r = 0; r < 8; ++r) acc[r] = bias;

#if __has_builtin(__builtin_amdgcn_wmma_f32_16x16x4_f32)
  #pragma unroll
  for (int k0 = 0; k0 < PD; k0 += 4) {
    v2f a  = *(const v2f*)(ap + k0);
    v2f bb = *(const v2f*)(wrow + k0);
    acc = __builtin_amdgcn_wmma_f32_16x16x4_f32(false, a, false, bb, (short)0, acc,
                                                false, false);
  }
#else
  // VALU fallback matching the C-matrix layout (M = r + 8*hi, N = l16)
  #pragma unroll
  for (int r = 0; r < 8; ++r) {
    const float* prow = pair + ((long long)bi * SEQ + j0 + r + 8 * hi) * PD;
    float s = 0.f;
    for (int k = 0; k < PD; ++k) s += prow[k] * wpm[l16 * PD + k];
    acc[r] += s;
  }
#endif

  // Transpose through LDS so stores to pb[b,h,i,j] are coalesced b128.
  __shared__ float sbuf[8][NH * 16];
  if (l16 < NH) {
    #pragma unroll
    for (int r = 0; r < 8; ++r) sbuf[wave][l16 * 16 + r + 8 * hi] = acc[r];
  }
  __syncthreads();
  const int hh = lane >> 2;          // head 0..7
  const int jj = (lane & 3) * 4;     // j offset 0,4,8,12
  v4f vv = *(const v4f*)&sbuf[wave][hh * 16 + jj];
  float* dst = pb + (((long long)(b * NH + hh) * SEQ + i) * SEQ + j0 + jj);
  *(v4f*)dst = vv;
}

// ---------------- Kernel 3: fused QKV projection (bf16 WMMA) ----------------
// out[m,n] = sum_k xb[m,k]*wqkv[n,k] + bias; q,k -> [B,H,S,64], v -> transposed [B,H,64,S]
__global__ __launch_bounds__(256) void qkv_gemm(
    const __bf16* __restrict__ xb, const __bf16* __restrict__ wqkv,
    const float* __restrict__ bq, const float* __restrict__ bk,
    const float* __restrict__ bv,
    __bf16* __restrict__ q_out, __bf16* __restrict__ k_out, __bf16* __restrict__ vt_out) {
  const int lane = threadIdx.x & 31, wave = threadIdx.x >> 5;
  const int l16 = lane & 15, hi = lane >> 4;
  const int tile = blockIdx.x * 8 + wave;          // (1536/16) * (1536/16) tiles
  const int tm = tile / (3 * EMB / 16);
  const int tn = tile % (3 * EMB / 16);

  const __bf16* arow = xb   + (long long)(tm * 16 + l16) * EMB;
  const __bf16* brow = wqkv + (long long)(tn * 16 + l16) * EMB;
  v8f acc = {};
  #pragma unroll
  for (int kc = 0; kc < EMB; kc += 32)
    acc = wmma_bf16(load_a16(arow, kc, hi), load_b16(brow, kc, hi), acc);

  const int ng  = tn * 16 + l16;
  const int mat = ng / EMB;            // 0=q 1=k 2=v (uniform per wave)
  const int nn  = ng % EMB;
  const float* bias = (mat == 0) ? bq : (mat == 1) ? bk : bv;
  const float bval = bias[nn];
  const int h = nn / HD, d = nn % HD;
  #pragma unroll
  for (int r = 0; r < 8; ++r) {
    const int mr = tm * 16 + r + 8 * hi;
    const int bb = mr / SEQ, s = mr % SEQ;
    const __bf16 val = (__bf16)(acc[r] + bval);
    const long long bh = (long long)(bb * NH + h);
    if (mat == 0)      q_out [(bh * SEQ + s) * HD + d] = val;
    else if (mat == 1) k_out [(bh * SEQ + s) * HD + d] = val;
    else               vt_out[(bh * HD + d) * SEQ + s] = val;
  }
}

// ---------------- Kernel 4: flash attention with pair bias, 4-way key split ----------
// Block = 8 waves = 2 query tiles x 4 key splits. Each wave computes unnormalized
// (m, l, ctx) over 192 keys; partials merged through LDS, each wave merging a 16-d slice.
__global__ __launch_bounds__(256) void attn_kernel(
    const __bf16* __restrict__ qb, const __bf16* __restrict__ kb,
    const __bf16* __restrict__ vtb, const float* __restrict__ pb,
    const unsigned char* __restrict__ mask, __bf16* __restrict__ ctxb) {
  const int lane = threadIdx.x & 31, wave = threadIdx.x >> 5;
  const int l16 = lane & 15, hi = lane >> 4;
  const int gp    = blockIdx.x * 2 + (wave >> 2);   // (bh, qtile) pair, 768 total
  const int split = wave & 3;
  const int qt = gp % (SEQ / 16);
  const int bh = gp / (SEQ / 16);
  const int b  = bh / NH, h = bh % NH;
  const int q0 = qt * 16;
  const float rsD = 0.125f;                          // 1/sqrt(64)

  __shared__ __bf16 plds[8][16 * 32];                // per-wave P tile (bf16)
  __shared__ float  ctxs[8][16][HD];                 // per-wave unnormalized ctx
  __shared__ float  marr[8][16], larr[8][16];        // per-wave softmax stats
  __bf16* pw = plds[wave];

  const __bf16* qrow = qb + ((long long)bh * SEQ + q0 + l16) * HD;
  const v16bf qa0 = load_a16(qrow, 0, hi);
  const v16bf qa1 = load_a16(qrow, 32, hi);

  v8f ctx0 = {}, ctx1 = {}, ctx2 = {}, ctx3 = {};
  float mrow[8], lrow[8];
  #pragma unroll
  for (int r = 0; r < 8; ++r) { mrow[r] = -1e30f; lrow[r] = 0.f; }

  const __bf16* kbase = kb + (long long)bh * SEQ * HD;
  const float*  pbase = pb + (long long)bh * SEQ * SEQ + (long long)q0 * SEQ;
  const unsigned char* mk = mask + b * SEQ;

  const int jbeg = split * KSEG;
  for (int j0 = jbeg; j0 < jbeg + KSEG; j0 += 32) {
    // ---- scores: Q(16x64) x K^T(64x32) via 4 WMMA ----
    v8f s0 = {}, s1 = {};
    {
      const __bf16* kr0 = kbase + (long long)(j0 + l16) * HD;
      s0 = wmma_bf16(qa0, load_b16(kr0, 0, hi), s0);
      s0 = wmma_bf16(qa1, load_b16(kr0, 32, hi), s0);
      const __bf16* kr1 = kbase + (long long)(j0 + 16 + l16) * HD;
      s1 = wmma_bf16(qa0, load_b16(kr1, 0, hi), s1);
      s1 = wmma_bf16(qa1, load_b16(kr1, 32, hi), s1);
    }
    const float m0 = mk[j0 + l16]      ? 0.f : -1e30f;
    const float m1 = mk[j0 + 16 + l16] ? 0.f : -1e30f;
    #pragma unroll
    for (int r = 0; r < 8; ++r) {
      const float* prow = pbase + (long long)(r + 8 * hi) * SEQ + j0;
      s0[r] = s0[r] * rsD + prow[l16] + m0;
      s1[r] = s1[r] * rsD + prow[16 + l16] + m1;
    }
    // ---- online softmax (row = r + 8*hi; reduce across 16-lane half) ----
    #pragma unroll
    for (int r = 0; r < 8; ++r) {
      float v = fmaxf(s0[r], s1[r]);
      #pragma unroll
      for (int off = 8; off >= 1; off >>= 1) v = fmaxf(v, __shfl_xor(v, off, 32));
      const float mnew = fmaxf(mrow[r], v);
      const float scale = __expf(mrow[r] - mnew);
      mrow[r] = mnew;
      const float p0 = __expf(s0[r] - mnew);
      const float p1 = __expf(s1[r] - mnew);
      float ps = p0 + p1;
      #pragma unroll
      for (int off = 8; off >= 1; off >>= 1) ps += __shfl_xor(ps, off, 32);
      lrow[r] = lrow[r] * scale + ps;
      ctx0[r] *= scale; ctx1[r] *= scale; ctx2[r] *= scale; ctx3[r] *= scale;
      const int qr = r + 8 * hi;
      pw[qr * 32 + l16]      = (__bf16)p0;
      pw[qr * 32 + 16 + l16] = (__bf16)p1;
    }
    __syncthreads();
    // ---- P(16x32) x V(32x64): A from LDS, B from transposed V (contiguous) ----
    {
      v8bf plo = *(const v8bf*)(pw + l16 * 32 + hi * 16);
      v8bf phi = *(const v8bf*)(pw + l16 * 32 + hi * 16 + 8);
      const v16bf pa = cat16(plo, phi);
      const __bf16* vr = vtb + ((long long)bh * HD + l16) * SEQ + j0 + hi * 16;
      v8bf a0, a1;
      a0 = *(const v8bf*)(vr);                  a1 = *(const v8bf*)(vr + 8);
      ctx0 = wmma_bf16(pa, cat16(a0, a1), ctx0);
      a0 = *(const v8bf*)(vr + 16LL * SEQ);     a1 = *(const v8bf*)(vr + 16LL * SEQ + 8);
      ctx1 = wmma_bf16(pa, cat16(a0, a1), ctx1);
      a0 = *(const v8bf*)(vr + 32LL * SEQ);     a1 = *(const v8bf*)(vr + 32LL * SEQ + 8);
      ctx2 = wmma_bf16(pa, cat16(a0, a1), ctx2);
      a0 = *(const v8bf*)(vr + 48LL * SEQ);     a1 = *(const v8bf*)(vr + 48LL * SEQ + 8);
      ctx3 = wmma_bf16(pa, cat16(a0, a1), ctx3);
    }
    __syncthreads();
  }

  // ---- publish unnormalized partials ----
  #pragma unroll
  for (int r = 0; r < 8; ++r) {
    const int row = r + 8 * hi;
    ctxs[wave][row][l16]       = ctx0[r];
    ctxs[wave][row][16 + l16]  = ctx1[r];
    ctxs[wave][row][32 + l16]  = ctx2[r];
    ctxs[wave][row][48 + l16]  = ctx3[r];
    if (l16 == 0) { marr[wave][row] = mrow[r]; larr[wave][row] = lrow[r]; }
  }
  __syncthreads();

  // ---- merge 4 splits; this wave handles its qtile's d-slice [split*16, +16) ----
  const int wq  = wave >> 2;           // which of the block's 2 qtiles
  const int dsl = split * 16;          // d-slice base
  #pragma unroll
  for (int rr = 0; rr < 8; ++rr) {
    const int row = rr + 8 * hi;
    float M = -1e30f;
    #pragma unroll
    for (int s = 0; s < 4; ++s) M = fmaxf(M, marr[wq * 4 + s][row]);
    float L = 0.f, C = 0.f;
    #pragma unroll
    for (int s = 0; s < 4; ++s) {
      const float e = __expf(marr[wq * 4 + s][row] - M);
      L += larr[wq * 4 + s][row] * e;
      C += ctxs[wq * 4 + s][row][dsl + l16] * e;
    }
    const float o = C / (L + 1e-20f);
    ctxb[((long long)b * SEQ + q0 + row) * EMB + h * HD + dsl + l16] = (__bf16)o;
  }
}

// ---------------- Kernel 5: output projection (bf16 WMMA, f32 out) ----------------
__global__ __launch_bounds__(256) void out_gemm(
    const __bf16* __restrict__ ctxb, const __bf16* __restrict__ wob,
    const float* __restrict__ bo, float* __restrict__ out) {
  const int lane = threadIdx.x & 31, wave = threadIdx.x >> 5;
  const int l16 = lane & 15, hi = lane >> 4;
  const int tile = blockIdx.x * 8 + wave;          // 96 x 32 tiles
  const int tm = tile / (EMB / 16);
  const int tn = tile % (EMB / 16);
  const __bf16* arow = ctxb + (long long)(tm * 16 + l16) * EMB;
  const __bf16* brow = wob  + (long long)(tn * 16 + l16) * EMB;
  v8f acc = {};
  #pragma unroll
  for (int kc = 0; kc < EMB; kc += 32)
    acc = wmma_bf16(load_a16(arow, kc, hi), load_b16(brow, kc, hi), acc);
  const float bval = bo[tn * 16 + l16];
  #pragma unroll
  for (int r = 0; r < 8; ++r)
    out[(long long)(tm * 16 + r + 8 * hi) * EMB + tn * 16 + l16] = acc[r] + bval;
}

// ---------------- Launcher ----------------
extern "C" void kernel_launch(void* const* d_in, const int* in_sizes, int n_in,
                              void* d_out, int out_size, void* d_ws, size_t ws_size,
                              hipStream_t stream) {
  (void)in_sizes; (void)n_in; (void)out_size; (void)ws_size;
  const float* x    = (const float*)d_in[0];
  const float* pair = (const float*)d_in[1];
  const unsigned char* mask = (const unsigned char*)d_in[2];
  const float* Wq = (const float*)d_in[3];
  const float* bq = (const float*)d_in[4];
  const float* Wk = (const float*)d_in[5];
  const float* bk = (const float*)d_in[6];
  const float* Wv = (const float*)d_in[7];
  const float* bv = (const float*)d_in[8];
  const float* Wo = (const float*)d_in[9];
  const float* bo = (const float*)d_in[10];
  const float* Wp = (const float*)d_in[11];
  const float* bp = (const float*)d_in[12];

  char* w = (char*)d_ws;
  size_t off = 0;
  auto alloc = [&](size_t bytes) {
    char* p = w + off; off += (bytes + 255) & ~(size_t)255; return p;
  };
  float*  pbuf = (float*) alloc((size_t)NBATCH * NH * SEQ * SEQ * sizeof(float)); // 37.7 MB
  __bf16* xb   = (__bf16*)alloc((size_t)NROW * EMB * 2);
  __bf16* wqkv = (__bf16*)alloc((size_t)3 * EMB * EMB * 2);
  __bf16* wob  = (__bf16*)alloc((size_t)EMB * EMB * 2);
  __bf16* qbuf = (__bf16*)alloc((size_t)NROW * EMB * 2);
  __bf16* kbuf = (__bf16*)alloc((size_t)NROW * EMB * 2);
  __bf16* vtb  = (__bf16*)alloc((size_t)NROW * EMB * 2);
  __bf16* ctxb = (__bf16*)alloc((size_t)NROW * EMB * 2);
  float*  wpm  = (float*) alloc((size_t)16 * PD * sizeof(float));   // padded Wp
  float*  bpad = (float*) alloc((size_t)16 * sizeof(float));        // padded bp

  const long long conv_elems =
      (long long)NROW * EMB + 4LL * EMB * EMB + 16 * PD + 16;
  convert_kernel<<<(int)((conv_elems + 255) / 256), 256, 0, stream>>>(
      x, Wq, Wk, Wv, Wo, Wp, bp, xb, wqkv, wob, wpm, bpad);
  pair_bias_kernel<<<NBATCH * SEQ * (SEQ / 128), 256, 0, stream>>>(pair, wpm, bpad, pbuf);
  qkv_gemm<<<(96 * 96) / 8, 256, 0, stream>>>(xb, wqkv, bq, bk, bv, qbuf, kbuf, vtb);
  attn_kernel<<<(NBATCH * NH * (SEQ / 16) * KSPLIT) / 8, 256, 0, stream>>>(
      qbuf, kbuf, vtb, pbuf, mask, ctxb);
  out_gemm<<<(96 * (EMB / 16)) / 8, 256, 0, stream>>>(ctxb, wob, bo, (float*)d_out);
}
